// SimpleNet_87720412054073
// MI455X (gfx1250) — matrix operand order repaired and checked by reading, not compiled, not run on hardware
//
#include <hip/hip_runtime.h>
#include <hip/hip_bf16.h>

// ---------------- CDNA5 WMMA types ----------------
typedef __attribute__((ext_vector_type(16))) _Float16 v16h;
typedef __attribute__((ext_vector_type(8)))  float    v8f;

#define HID   128
#define KP1   96     // padded K for layer 1 (68 -> 96 = 3 * 32)
#define NCOEF 6
#define IN_DIM 8
#define BATCH 2048
#define ABUD  512
#define XCOLS 39

// ---- LDS layout (units: _Float16 elements unless noted) ----
#define OFF_WF1  0                         // [128][96]
#define OFF_WF2  (OFF_WF1 + HID*KP1)       // [128][128]
#define OFF_WC1  (OFF_WF2 + HID*HID)       // [128][128]
#define OFF_WB1  (OFF_WC1 + HID*HID)       // [128][128]
#define OFF_WC2  (OFF_WB1 + HID*HID)       // [48][128]
#define OFF_ACT  (OFF_WC2 + 48*HID)        // per wave: 2 buffers of 16*128
#define ACT_HALVES_PER_WAVE (2*16*HID)
#define OFF_COEF_BYTES ((OFF_ACT + 4*ACT_HALVES_PER_WAVE) * 2)   // f32 coef region
#define COEF_FLOATS_PER_WAVE (16*48)
#define OFF_WB2_BYTES (OFF_COEF_BYTES + 4*COEF_FLOATS_PER_WAVE*4) // Wb2 as f32[128]
#define SMEM_BYTES (OFF_WB2_BYTES + HID*4)

__device__ inline void wait_ds_all() {
    asm volatile("s_wait_dscnt 0" ::: "memory");
}

// Stage global f32 weight (K x N, row-major) into LDS as f16 transposed [N][Kp],
// zero-padding K -> Kp.  All threads of the block cooperate.
__device__ inline void stage_wT(_Float16* dst, const float* W, int K, int N, int Kp,
                                int tid, int nthreads) {
    for (int i = tid; i < N * Kp; i += nthreads) {
        int n = i / Kp, k = i - n * Kp;
        float v = (k < K) ? W[k * N + n] : 0.f;
        dst[n * Kp + k] = (_Float16)v;
    }
}

// A fragment: 16xK tile (row-major in LDS, stride Kp), ISA 16-bit A layout:
// lane L: row m = L&15, hi = L>>4 selects K-offset +8 within each 16-K group.
// VGPR j<4 -> K = 2j(+1), VGPR j>=4 -> K = 16+2(j-4)(+1), both + hi*8 + ksBase.
// Per lane this is two contiguous 16-byte runs -> compiler emits 2x ds_load_b128.
__device__ inline v16h load_a(const _Float16* act, int Kp, int ksBase, int lane) {
    v16h a;
    int m = lane & 15, hi = lane >> 4;
#pragma unroll
    for (int j = 0; j < 8; ++j) {
        int k0 = ksBase + hi * 8 + ((j < 4) ? 2 * j : 16 + 2 * (j - 4));
        a[2 * j]     = act[m * Kp + k0];
        a[2 * j + 1] = act[m * Kp + k0 + 1];
    }
    return a;
}

// B fragment: KxN weight stored transposed [N][Kp] in LDS.  ISA 16-bit B layout:
// lane L: col n = ntBase + (L&15), hi = L>>4 selects K-offset +8 within each 16-K group;
// VGPR j<4 -> K = 2j(+1), VGPR j>=4 -> K = 16+2(j-4)(+1), both + hi*8 + ksBase.
// Two contiguous 16-byte runs per lane -> 2x ds_load_b128.
__device__ inline v16h load_b(const _Float16* wt, int Kp, int ksBase, int ntBase, int lane) {
    v16h b;
    int n = ntBase + (lane & 15), hi = lane >> 4;
#pragma unroll
    for (int j = 0; j < 8; ++j) {
        int k0 = ksBase + hi * 8 + ((j < 4) ? 2 * j : 16 + 2 * (j - 4));
        b[2 * j]     = wt[n * Kp + k0];
        b[2 * j + 1] = wt[n * Kp + k0 + 1];
    }
    return b;
}

// One dense layer for a single wave's 16-row tile: (16 x K) @ (K x N) + bias [+relu].
// A fragments are hoisted out of the N-tile loop (invariant across nt) -> inner loop
// only streams B tiles from LDS.  C/D layout: lane = col (lo/hi), VGPR r = row r+hi*8.
template <int KP, int KTILES, int N, bool RELU>
__device__ inline void wmma_layer_h(const _Float16* actIn, const _Float16* wt,
                                    const float* bias, _Float16* actOut, int outStride,
                                    int lane) {
    int n = lane & 15, hi = lane >> 4;
    v16h afrag[KTILES];
#pragma unroll
    for (int ks = 0; ks < KTILES; ++ks) afrag[ks] = load_a(actIn, KP, ks * 32, lane);
    for (int nt = 0; nt < N; nt += 16) {
        v8f acc = {};
#pragma unroll
        for (int ks = 0; ks < KTILES; ++ks) {
            v16h b = load_b(wt, KP, ks * 32, nt, lane);
            acc = __builtin_amdgcn_wmma_f32_16x16x32_f16(
                false, afrag[ks], false, b, (short)0, acc, false, false);
        }
        float bv = bias[nt + n];
#pragma unroll
        for (int r = 0; r < 8; ++r) {
            float v = acc[r] + bv;
            if (RELU) v = fmaxf(v, 0.f);
            actOut[(r + hi * 8) * outStride + nt + n] = (_Float16)v;
        }
    }
    wait_ds_all();
}

// Same, but f32 output (final coef head, no relu).
template <int KP, int KTILES, int N>
__device__ inline void wmma_layer_f(const _Float16* actIn, const _Float16* wt,
                                    const float* bias, float* outF, int outStride,
                                    int lane) {
    int n = lane & 15, hi = lane >> 4;
    v16h afrag[KTILES];
#pragma unroll
    for (int ks = 0; ks < KTILES; ++ks) afrag[ks] = load_a(actIn, KP, ks * 32, lane);
    for (int nt = 0; nt < N; nt += 16) {
        v8f acc = {};
#pragma unroll
        for (int ks = 0; ks < KTILES; ++ks) {
            v16h b = load_b(wt, KP, ks * 32, nt, lane);
            acc = __builtin_amdgcn_wmma_f32_16x16x32_f16(
                false, afrag[ks], false, b, (short)0, acc, false, false);
        }
        float bv = bias[nt + n];
#pragma unroll
        for (int r = 0; r < 8; ++r)
            outF[(r + hi * 8) * outStride + nt + n] = acc[r] + bv;
    }
    wait_ds_all();
}

// ============ Kernel 1: fused embed + MLP, emits per-sample {cbar[6], scale} ============
// grid = 32 blocks x 128 threads (4 waves); each wave owns 16 batch rows.
__global__ void kan_mlp_kernel(const float* __restrict__ x,
                               const float* __restrict__ emb_id,
                               const float* __restrict__ emb_period,
                               const float* __restrict__ emb_time,
                               const float* __restrict__ Wf1, const float* __restrict__ bf1,
                               const float* __restrict__ Wf2, const float* __restrict__ bf2,
                               const float* __restrict__ Wc1, const float* __restrict__ bc1,
                               const float* __restrict__ Wc2, const float* __restrict__ bc2,
                               const float* __restrict__ Wb1, const float* __restrict__ bb1,
                               const float* __restrict__ Wb2, const float* __restrict__ bb2,
                               float* __restrict__ ws) {
    extern __shared__ __align__(16) unsigned char smem_raw[];
    _Float16* lds = (_Float16*)smem_raw;
    float* ldsWb2 = (float*)(smem_raw + OFF_WB2_BYTES);

    const int tid = threadIdx.x;
    const int nthreads = blockDim.x;

    // ---- cooperatively stage all weights (f32 -> f16, transposed [N][Kp]) ----
    stage_wT(lds + OFF_WF1, Wf1,  68, HID, KP1, tid, nthreads);
    stage_wT(lds + OFF_WF2, Wf2, HID, HID, HID, tid, nthreads);
    stage_wT(lds + OFF_WC1, Wc1, HID, HID, HID, tid, nthreads);
    stage_wT(lds + OFF_WB1, Wb1, HID, HID, HID, tid, nthreads);
    stage_wT(lds + OFF_WC2, Wc2, HID,  48, HID, tid, nthreads);
    if (tid < HID) ldsWb2[tid] = Wb2[tid];    // keep f32 precision for the scalar head
    __syncthreads();

    const int wave = tid >> 5;
    const int lane = tid & 31;
    const int rowBase = blockIdx.x * 64 + wave * 16;

    _Float16* actA = lds + OFF_ACT + wave * ACT_HALVES_PER_WAVE;          // 16x128
    _Float16* actB = actA + 16 * HID;                                     // 16x128
    float* coefBuf = (float*)(smem_raw + OFF_COEF_BYTES) + wave * COEF_FLOATS_PER_WAVE; // 16x48

    // ---- per-wave embedding gather into actA as 16 x 96 (padded) f16 ----
    for (int i = lane; i < 16 * KP1; i += 32) {
        int r = i / KP1, c = i - r * KP1;
        const float* xr = x + (size_t)(rowBase + r) * XCOLS;
        float v;
        if (c < 8)        { int ii = (int)xr[1]; v = emb_id[ii * 8 + c]; }
        else if (c < 16)  { int ip = (int)xr[0]; v = emb_period[ip * 8 + (c - 8)]; }
        else if (c < 32)  { int it = (int)xr[2]; v = emb_time[it * 16 + (c - 16)]; }
        else if (c < 68)  { v = xr[3 + (c - 32)]; }
        else              { v = 0.f; }
        actA[r * KP1 + c] = (_Float16)v;
    }
    wait_ds_all();

    // ---- MLP chain (each wave fully independent after the weight barrier) ----
    // h1 = relu(emb @ Wf1 + bf1)        (16x96)@(96x128)
    wmma_layer_h<KP1, 3, HID, true>(actA, lds + OFF_WF1, bf1, actB, HID, lane);
    // h  = relu(h1 @ Wf2 + bf2)         (16x128)@(128x128)
    wmma_layer_h<HID, 4, HID, true>(actB, lds + OFF_WF2, bf2, actA, HID, lane);
    // hc = relu(h @ Wc1 + bc1)
    wmma_layer_h<HID, 4, HID, true>(actA, lds + OFF_WC1, bc1, actB, HID, lane);
    // coef = hc @ Wc2 + bc2             (16x128)@(128x48) -> f32
    wmma_layer_f<HID, 4, 48>(actB, lds + OFF_WC2, bc2, coefBuf, 48, lane);
    // hb = relu(h @ Wb1 + bb1)          (reuses actB; coef already extracted)
    wmma_layer_h<HID, 4, HID, true>(actA, lds + OFF_WB1, bb1, actB, HID, lane);

    // ---- scalar heads: scale_base = hb . Wb2 + bb2 ; cbar = mean_j coef ----
    if (lane < 16) {
        float s = bb2[0];
        for (int k = 0; k < HID; ++k)
            s += (float)actB[lane * HID + k] * ldsWb2[k];
        float* wrow = ws + (size_t)(rowBase + lane) * 8;
        wrow[6] = s;
#pragma unroll
        for (int k = 0; k < NCOEF; ++k) {
            float acc = 0.f;
#pragma unroll
            for (int j = 0; j < IN_DIM; ++j)
                acc += coefBuf[lane * 48 + j * NCOEF + k];
            wrow[k] = acc * (1.f / IN_DIM);
        }
    }
}

// ============ Kernel 2: per-point uniform cubic B-spline + silu + scale ============
// One block per batch row. out[b,a] = scale * ( silu(u) + sum_k B_k(u) * cbar[k] )
__global__ void kan_spline_kernel(const float* __restrict__ budget,
                                  const float* __restrict__ ws,
                                  float* __restrict__ out) {
    const int b = blockIdx.x;
    __shared__ float p[8];
    if (threadIdx.x < 8) p[threadIdx.x] = ws[(size_t)b * 8 + threadIdx.x];
    __syncthreads();

    // extended knot vector: linspace(-1,1,4) extended by k=3 on each side (uniform, h=2/3)
    const float t0 = -3.f, h = 2.f / 3.f;
    float t[10];
#pragma unroll
    for (int i = 0; i < 10; ++i) t[i] = t0 + h * i;

    for (int a = threadIdx.x; a < ABUD; a += blockDim.x) {
        float u = budget[(size_t)b * ABUD + a];
        float B0[9], B1[8], B2[7], B3[6];
#pragma unroll
        for (int i = 0; i < 9; ++i) B0[i] = (u >= t[i] && u < t[i + 1]) ? 1.f : 0.f;
#pragma unroll
        for (int i = 0; i < 8; ++i)
            B1[i] = (u - t[i]) / (t[i + 1] - t[i]) * B0[i]
                  + (t[i + 2] - u) / (t[i + 2] - t[i + 1]) * B0[i + 1];
#pragma unroll
        for (int i = 0; i < 7; ++i)
            B2[i] = (u - t[i]) / (t[i + 2] - t[i]) * B1[i]
                  + (t[i + 3] - u) / (t[i + 3] - t[i + 1]) * B1[i + 1];
#pragma unroll
        for (int i = 0; i < 6; ++i)
            B3[i] = (u - t[i]) / (t[i + 3] - t[i]) * B2[i]
                  + (t[i + 4] - u) / (t[i + 4] - t[i + 1]) * B2[i + 1];
        float spl = 0.f;
#pragma unroll
        for (int k = 0; k < NCOEF; ++k) spl += B3[k] * p[k];
        float sig = 1.f / (1.f + __expf(-u));
        out[(size_t)b * ABUD + a] = p[6] * (u * sig + spl);
    }
}

extern "C" void kernel_launch(void* const* d_in, const int* in_sizes, int n_in,
                              void* d_out, int out_size, void* d_ws, size_t ws_size,
                              hipStream_t stream) {
    const float* x          = (const float*)d_in[0];
    const float* budget     = (const float*)d_in[1];
    const float* emb_id     = (const float*)d_in[2];
    const float* emb_period = (const float*)d_in[3];
    const float* emb_time   = (const float*)d_in[4];
    const float* Wf1 = (const float*)d_in[5];  const float* bf1 = (const float*)d_in[6];
    const float* Wf2 = (const float*)d_in[7];  const float* bf2 = (const float*)d_in[8];
    const float* Wc1 = (const float*)d_in[9];  const float* bc1 = (const float*)d_in[10];
    const float* Wc2 = (const float*)d_in[11]; const float* bc2 = (const float*)d_in[12];
    // d_in[13..16] = Ws1/bs1/Ws2/bs2 : computed-but-unused in the reference -> skipped
    const float* Wb1 = (const float*)d_in[17]; const float* bb1 = (const float*)d_in[18];
    const float* Wb2 = (const float*)d_in[19]; const float* bb2 = (const float*)d_in[20];

    float* ws  = (float*)d_ws;           // BATCH x 8 floats: cbar[0..5], scale, pad
    float* out = (float*)d_out;          // BATCH x ABUD

    kan_mlp_kernel<<<dim3(BATCH / 64), dim3(128), SMEM_BYTES, stream>>>(
        x, emb_id, emb_period, emb_time,
        Wf1, bf1, Wf2, bf2, Wc1, bc1, Wc2, bc2, Wb1, bb1, Wb2, bb2, ws);

    kan_spline_kernel<<<dim3(BATCH), dim3(256), 0, stream>>>(budget, ws, out);
}
